// AttentionHead_42520176231029
// MI455X (gfx1250) — matrix operand order repaired and checked by reading, not compiled, not run on hardware
//
#include <hip/hip_runtime.h>

#define BB 4
#define SS 4096
#define DD 256

typedef __bf16 bf16_t;
typedef __attribute__((ext_vector_type(16))) __bf16 v16bf;
typedef __attribute__((ext_vector_type(8)))  __bf16 v8bf;
typedef __attribute__((ext_vector_type(8)))  float  v8f;

__device__ __forceinline__ v16bf join8(v8bf lo, v8bf hi) {
  return __builtin_shufflevector(lo, hi, 0,1,2,3,4,5,6,7,8,9,10,11,12,13,14,15);
}

// 16 contiguous f32 -> v16bf (WMMA B operand: element i = K index i)
__device__ __forceinline__ v16bf load_cvt16(const float* __restrict__ p) {
  v16bf r;
#pragma unroll
  for (int i = 0; i < 16; ++i) r[i] = (__bf16)p[i];
  return r;
}

// WMMA A operand halves: elements 0..7 from lo, 8..15 from hi
__device__ __forceinline__ v16bf cvt_split16(const float* __restrict__ lo,
                                             const float* __restrict__ hi) {
  v16bf r;
#pragma unroll
  for (int i = 0; i < 8; ++i) r[i] = (__bf16)lo[i];
#pragma unroll
  for (int i = 0; i < 8; ++i) r[8 + i] = (__bf16)hi[i];
  return r;
}

__device__ __forceinline__ v8f wmma_bf16(v16bf a, v16bf b, v8f c) {
  return __builtin_amdgcn_wmma_f32_16x16x32_bf16(false, a, false, b, (short)0, c, false, false);
}

// ---------------------------------------------------------------------------
// Y = X @ W^T  (torch Linear), X:[B,S,D] f32, W:[D,D] f32 -> Y:[B,S,D] bf16
// Computed as Y^T tiles: A = W (M=e,K=d), B = X^T (K=d,N=s) so the D-tile
// stores map to contiguous row-major bf16 writes of Y.
// Wave: 16 e-rows x 64 s-cols. 1024 waves per batch.
// ---------------------------------------------------------------------------
__global__ void __launch_bounds__(256)
proj_qk_kernel(const float* __restrict__ X, const float* __restrict__ W,
               bf16_t* __restrict__ Y) {
  const int lane = threadIdx.x & 31;
  const int wv   = threadIdx.x >> 5;
  const int widx = blockIdx.x * 8 + wv;
  const int bb   = blockIdx.y;
  const int nl   = lane & 15;
  const int h    = lane >> 4;
  const int e0   = (widx >> 6) * 16;   // 16 e-tiles
  const int sg   = (widx & 63) * 64;   // 64 s-groups of 64

  const float* Xb = X + (size_t)bb * SS * DD;
  v8f acc[4];
#pragma unroll
  for (int nt = 0; nt < 4; ++nt) acc[nt] = {0.f,0.f,0.f,0.f,0.f,0.f,0.f,0.f};

  for (int kk = 0; kk < 8; ++kk) {
    const int d0 = kk * 32;
    const float* wrow = W + (size_t)(e0 + nl) * DD + d0;
    v16bf a = cvt_split16(wrow + h * 16, wrow + h * 16 + 8);
#pragma unroll
    for (int nt = 0; nt < 4; ++nt) {
      const float* xrow = Xb + (size_t)(sg + nt * 16 + nl) * DD + d0 + h * 16;
      acc[nt] = wmma_bf16(a, load_cvt16(xrow), acc[nt]);
    }
  }
  bf16_t* Yb = Y + (size_t)bb * SS * DD;
#pragma unroll
  for (int nt = 0; nt < 4; ++nt) {
    v8bf pk;
#pragma unroll
    for (int v = 0; v < 8; ++v) pk[v] = (__bf16)acc[nt][v];
    *(v8bf*)(Yb + (size_t)(sg + nt * 16 + nl) * DD + e0 + 8 * h) = pk;
  }
}

// ---------------------------------------------------------------------------
// Vt = (X @ W^T)^T stored as [B, D, S] bf16 (S contiguous) so the attention
// kernel loads V directly in WMMA B layout.  A = X (M=s,K=d), B[d,e]=W[e,d].
// Wave: 16 s-rows x 64 e-cols. 1024 waves per batch.
// ---------------------------------------------------------------------------
__global__ void __launch_bounds__(256)
proj_v_kernel(const float* __restrict__ X, const float* __restrict__ W,
              bf16_t* __restrict__ Vt) {
  const int lane = threadIdx.x & 31;
  const int wv   = threadIdx.x >> 5;
  const int widx = blockIdx.x * 8 + wv;
  const int bb   = blockIdx.y;
  const int nl   = lane & 15;
  const int h    = lane >> 4;
  const int s0   = (widx >> 2) * 16;   // 256 s-tiles
  const int eg   = (widx & 3) * 64;    // 4 e-groups of 64

  const float* Xb = X + (size_t)bb * SS * DD;
  v8f acc[4];
#pragma unroll
  for (int nt = 0; nt < 4; ++nt) acc[nt] = {0.f,0.f,0.f,0.f,0.f,0.f,0.f,0.f};

  for (int kk = 0; kk < 8; ++kk) {
    const int d0 = kk * 32;
    const float* xrow = Xb + (size_t)(s0 + nl) * DD + d0;
    v16bf a = cvt_split16(xrow + h * 16, xrow + h * 16 + 8);
#pragma unroll
    for (int nt = 0; nt < 4; ++nt) {
      const float* wrow = W + (size_t)(eg + nt * 16 + nl) * DD + d0 + h * 16;
      acc[nt] = wmma_bf16(a, load_cvt16(wrow), acc[nt]);
    }
  }
  bf16_t* Vb = Vt + (size_t)bb * DD * SS;
#pragma unroll
  for (int nt = 0; nt < 4; ++nt) {
    v8bf pk;
#pragma unroll
    for (int v = 0; v < 8; ++v) pk[v] = (__bf16)acc[nt][v];
    *(v8bf*)(Vb + (size_t)(eg + nt * 16 + nl) * SS + s0 + 8 * h) = pk;
  }
}

// ---------------------------------------------------------------------------
// Flash attention forward.  Q,K:[B,S,D] bf16 row-major; Vt:[B,D,S] bf16.
// Block = 8 waves; wave w owns query rows q0..q0+15 independently.
// Streaming softmax over 64-key blocks (4 independent QK^T accumulators),
// causal loop bound.  D-layout convention: row = v + 8*h, col = lane%16.
// ---------------------------------------------------------------------------
__global__ void __launch_bounds__(256)
flash_attn_kernel(const bf16_t* __restrict__ Q, const bf16_t* __restrict__ K,
                  const bf16_t* __restrict__ Vt, const int* __restrict__ amask,
                  float* __restrict__ out) {
  __shared__ __align__(32) bf16_t plds[8][16 * 64];  // per-wave P scratch (16 rows x 64 keys)

  const int lane = threadIdx.x & 31;
  const int wv   = threadIdx.x >> 5;
  const int nl   = lane & 15;
  const int h    = lane >> 4;
  const int bb   = blockIdx.y;
  const int q0   = blockIdx.x * 128 + wv * 16;

  // Q rows into WMMA A-layout registers: lane nl -> row q0+nl
  v16bf qreg[8];
  {
    const bf16_t* qbase = Q + ((size_t)bb * SS + q0 + nl) * DD;
#pragma unroll
    for (int kk = 0; kk < 8; ++kk) {
      v8bf lo = *(const v8bf*)(qbase + kk * 32 + h * 16);
      v8bf hi = *(const v8bf*)(qbase + kk * 32 + h * 16 + 8);
      qreg[kk] = join8(lo, hi);
    }
  }

  v8f o[16];
#pragma unroll
  for (int dt = 0; dt < 16; ++dt) o[dt] = {0.f,0.f,0.f,0.f,0.f,0.f,0.f,0.f};
  float m_st[8], l_st[8];
#pragma unroll
  for (int v = 0; v < 8; ++v) { m_st[v] = -3.0e38f; l_st[v] = 0.f; }

  const float sc = 0.0625f;                 // 1/sqrt(256)
  const int nblk = (q0 + 15) / 64 + 1;      // causal: only kb <= q0+15
  const bf16_t* Kb = K  + (size_t)bb * SS * DD;
  const bf16_t* Vb = Vt + (size_t)bb * DD * SS;
  const int*    mb = amask + (size_t)bb * SS;

  for (int ib = 0; ib < nblk; ++ib) {
    const int kb = ib * 64;

    // ---- S = Q K^T for 64 keys: 4 independent 16-wide accumulators ----
    v8f sacc[4];
#pragma unroll
    for (int nt = 0; nt < 4; ++nt) sacc[nt] = {0.f,0.f,0.f,0.f,0.f,0.f,0.f,0.f};
#pragma unroll
    for (int kk = 0; kk < 8; ++kk) {
#pragma unroll
      for (int nt = 0; nt < 4; ++nt) {
        v16bf bt = *(const v16bf*)(Kb + (size_t)(kb + nt * 16 + nl) * DD + kk * 32 + h * 16);
        sacc[nt] = wmma_bf16(qreg[kk], bt, sacc[nt]);
      }
    }

    // ---- masks + streaming softmax ----
    int kidx[4], pm[4];
#pragma unroll
    for (int nt = 0; nt < 4; ++nt) {
      kidx[nt] = kb + nt * 16 + nl;
      pm[nt]   = mb[kidx[nt]];
    }
    float alpha[8], p[4][8];
#pragma unroll
    for (int v = 0; v < 8; ++v) {
      const int q = q0 + v + 8 * h;
      float x[4];
#pragma unroll
      for (int nt = 0; nt < 4; ++nt) {
        const bool ok = (kidx[nt] <= q) && (pm[nt] != 0);
        x[nt] = ok ? sacc[nt][v] * sc : -3.0e38f;
      }
      float t = fmaxf(fmaxf(x[0], x[1]), fmaxf(x[2], x[3]));
      t = fmaxf(t, __shfl_xor(t, 1, 32));
      t = fmaxf(t, __shfl_xor(t, 2, 32));
      t = fmaxf(t, __shfl_xor(t, 4, 32));
      t = fmaxf(t, __shfl_xor(t, 8, 32));
      const float mnew = fmaxf(m_st[v], t);
      alpha[v] = __expf(m_st[v] - mnew);
      float rs = 0.f;
#pragma unroll
      for (int nt = 0; nt < 4; ++nt) {
        // x[nt] == -3e38 for masked entries -> p forced to 0, exp skipped
        const float e = (x[nt] > -1.0e38f) ? __expf(x[nt] - mnew) : 0.f;
        p[nt][v] = e;
        rs += e;
      }
      rs += __shfl_xor(rs, 1, 32);
      rs += __shfl_xor(rs, 2, 32);
      rs += __shfl_xor(rs, 4, 32);
      rs += __shfl_xor(rs, 8, 32);
      l_st[v] = l_st[v] * alpha[v] + rs;
      m_st[v] = mnew;
    }

    // rescale running O
#pragma unroll
    for (int dt = 0; dt < 16; ++dt)
#pragma unroll
      for (int v = 0; v < 8; ++v) o[dt][v] *= alpha[v];

    // ---- transpose P (D-layout -> A-layout) via per-wave LDS scratch ----
#pragma unroll
    for (int v = 0; v < 8; ++v) {
      const int row = v + 8 * h;
#pragma unroll
      for (int nt = 0; nt < 4; ++nt)
        plds[wv][row * 64 + nt * 16 + nl] = (__bf16)p[nt][v];
    }
    // same-wave DS ops complete in order; read back in A-layout (two k-steps)
    v16bf pa0, pa1;
    {
      v8bf lo0 = *(const v8bf*)&plds[wv][nl * 64 + h * 16];
      v8bf hi0 = *(const v8bf*)&plds[wv][nl * 64 + h * 16 + 8];
      v8bf lo1 = *(const v8bf*)&plds[wv][nl * 64 + 32 + h * 16];
      v8bf hi1 = *(const v8bf*)&plds[wv][nl * 64 + 32 + h * 16 + 8];
      pa0 = join8(lo0, hi0);
      pa1 = join8(lo1, hi1);
    }

    // ---- O += P @ V  (Vt supplies B operand directly; 2 k-steps of 32) ----
#pragma unroll
    for (int dt = 0; dt < 16; ++dt) {
      const bf16_t* vrow = Vb + (size_t)(dt * 16 + nl) * SS + kb + h * 16;
      v16bf bv0 = *(const v16bf*)(vrow);
      v16bf bv1 = *(const v16bf*)(vrow + 32);
      o[dt] = wmma_bf16(pa0, bv0, o[dt]);
      o[dt] = wmma_bf16(pa1, bv1, o[dt]);
    }
  }

  // ---- epilogue: O / l, store f32 ----
  float inv_l[8];
#pragma unroll
  for (int v = 0; v < 8; ++v) inv_l[v] = 1.0f / l_st[v];
  float* ob = out + ((size_t)bb * SS + q0) * DD;
#pragma unroll
  for (int dt = 0; dt < 16; ++dt)
#pragma unroll
    for (int v = 0; v < 8; ++v)
      ob[(size_t)(v + 8 * h) * DD + dt * 16 + nl] = o[dt][v] * inv_l[v];
}

extern "C" void kernel_launch(void* const* d_in, const int* in_sizes, int n_in,
                              void* d_out, int out_size, void* d_ws, size_t ws_size,
                              hipStream_t stream) {
  const float* x_q = (const float*)d_in[0];
  const float* x_k = (const float*)d_in[1];
  const float* x_v = (const float*)d_in[2];
  const int*   am  = (const int*)d_in[3];
  const float* Wq  = (const float*)d_in[4];
  const float* Wk  = (const float*)d_in[5];
  const float* Wv  = (const float*)d_in[6];
  float* out = (float*)d_out;

  const size_t mat = (size_t)BB * SS * DD;   // elements per bf16 matrix
  bf16_t* Qbf  = (bf16_t*)d_ws;
  bf16_t* Kbf  = Qbf + mat;
  bf16_t* Vtbf = Kbf + mat;                  // total ws use: 24 MiB

  dim3 blk(256);
  dim3 gproj(128, BB);                       // 1024 waves per batch
  hipLaunchKernelGGL(proj_qk_kernel, gproj, blk, 0, stream, x_q, Wq, Qbf);
  hipLaunchKernelGGL(proj_qk_kernel, gproj, blk, 0, stream, x_k, Wk, Kbf);
  hipLaunchKernelGGL(proj_v_kernel,  gproj, blk, 0, stream, x_v, Wv, Vtbf);

  dim3 gfa(SS / 128, BB);                    // 32 q-tiles x 4 batches
  hipLaunchKernelGGL(flash_attn_kernel, gfa, blk, 0, stream, Qbf, Kbf, Vtbf, am, out);
}
